// MolGIN_70300024701668
// MI455X (gfx1250) — compile-verified, matrix operand order checked by reading, not run on hardware
//
#include <hip/hip_runtime.h>

#define N_NODES 100000
#define N_EDGES 300000
#define N_GRAPHS 2048
#define IN_DIM 38
#define HID 256
#define LAYERS 5

typedef __attribute__((ext_vector_type(16))) __bf16 v16bf;
typedef __attribute__((ext_vector_type(8)))  __bf16 v8bf;
typedef __attribute__((ext_vector_type(8)))  float  v8f;

static __device__ __forceinline__ __bf16 f2bf(float f) {
    unsigned u = __builtin_bit_cast(unsigned, f);
    unsigned r = u + 0x7FFFu + ((u >> 16) & 1u);   // round to nearest even
    unsigned short h = (unsigned short)(r >> 16);
    return __builtin_bit_cast(__bf16, h);
}

// Load one 16-row x 32-K bf16 WMMA fragment for this lane.
// p must already include the lane's kgrp offset; layout per CDNA5 ISA:
// lanes 0-15 hold K {0..7,16..23}, lanes 16-31 hold K {8..15,24..31}.
static __device__ __forceinline__ v16bf load_frag(const __bf16* p) {
    v8bf lo = *(const v8bf*)(p);
    v8bf hi = *(const v8bf*)(p + 16);
    v16bf f;
#pragma unroll
    for (int i = 0; i < 8; ++i) { f[i] = lo[i]; f[i + 8] = hi[i]; }
    return f;
}

// ---------------- projection: h = x @ Wp + bp  (K=38, memory bound, f32) ----
__global__ __launch_bounds__(256) void proj_kernel(
    const float* __restrict__ x, const float* __restrict__ Wp,
    const float* __restrict__ bp, float* __restrict__ h) {
    __shared__ float xs[4][IN_DIM];
    int row0 = blockIdx.x * 4;
    int c = threadIdx.x;
    for (int i = threadIdx.x; i < 4 * IN_DIM; i += 256) {
        int r = i / IN_DIM, k = i % IN_DIM;
        xs[r][k] = x[(size_t)(row0 + r) * IN_DIM + k];
    }
    __syncthreads();
    for (int r = 0; r < 4; ++r) {
        float acc = bp[c];
#pragma unroll
        for (int k = 0; k < IN_DIM; ++k) acc += xs[r][k] * Wp[k * HID + c];
        h[(size_t)(row0 + r) * HID + c] = acc;
    }
}

// ---------------- z = (1+eps_l) * h ----------------
__global__ __launch_bounds__(256) void gin_init_kernel(
    const float* __restrict__ h, float* __restrict__ z,
    const float* __restrict__ eps, int layer) {
    size_t i = (size_t)blockIdx.x * 256 + threadIdx.x;
    z[i] = (1.0f + eps[layer]) * h[i];
}

// ---------------- scatter-add over edges: z[dst] += h[src] ----------------
__global__ __launch_bounds__(256) void edge_scatter_kernel(
    const int* __restrict__ ei, const float* __restrict__ h,
    float* __restrict__ z) {
    int e = blockIdx.x;
    int src = ei[e];
    int dst = ei[N_EDGES + e];
    int c = threadIdx.x;
    atomicAdd(&z[(size_t)dst * HID + c], h[(size_t)src * HID + c]);
}

// ---------------- f32 -> bf16 ----------------
__global__ __launch_bounds__(256) void cvt_bf16_kernel(
    const float* __restrict__ in, __bf16* __restrict__ out) {
    size_t i = (size_t)blockIdx.x * 256 + threadIdx.x;
    out[i] = f2bf(in[i]);
}

// ---------------- transpose-convert weights: WT[n][k] = bf16(W[k][n]) ------
__global__ __launch_bounds__(256) void wconvT_kernel(
    const float* __restrict__ W, __bf16* __restrict__ WT) {
    int k = blockIdx.x;                   // 0..255
    int n = threadIdx.x;                  // 0..255
    WT[(size_t)n * HID + k] = f2bf(W[(size_t)k * HID + n]);
}

// ---------------- bf16 WMMA GEMM: C[M,256] = A[M,256] @ B + bias ----------
// A row-major bf16, BT = B^T row-major bf16 ([N][K]).
// One wave computes a 32(M) x 64(N) tile: 2 A-frags, 4 B-frags, 8 v8f accs.
// All fragment loads for a k-step are issued before the 8 WMMAs so the
// scheduler can use partial s_wait_loadcnt and overlap math with loads.
template <bool RELU, bool OUTBF>
__global__ __launch_bounds__(256) void gemm_bf16_wmma(
    const __bf16* __restrict__ A, const __bf16* __restrict__ BT,
    const float* __restrict__ bias, float* __restrict__ Cf,
    __bf16* __restrict__ Cb, int mtiles32) {
    const int K = HID, N = HID;
    int wave = (int)((blockIdx.x * 256 + threadIdx.x) >> 5);
    int lane = threadIdx.x & 31;
    int tile_m = wave >> 2;               // N/64 = 4 column tiles
    int tile_n = wave & 3;
    if (tile_m >= mtiles32) return;       // wave-uniform exit (EXEC stays full)

    int lmod = lane & 15;
    int kgrp = (lane >> 4) * 8;           // ISA 16-bit frag K grouping

    const __bf16* arow0 = A + (size_t)(tile_m * 32 + lmod) * K + kgrp;
    const __bf16* arow1 = arow0 + (size_t)16 * K;
    const __bf16* brow  = BT + (size_t)(tile_n * 64 + lmod) * K + kgrp;

    // warm B tile into caches (global_prefetch_b8)
#pragma unroll
    for (int j = 0; j < 4; ++j)
        __builtin_prefetch(brow + (size_t)j * 16 * K, 0, 3);

    v8f acc[2][4];
#pragma unroll
    for (int h = 0; h < 2; ++h)
#pragma unroll
        for (int j = 0; j < 4; ++j)
#pragma unroll
            for (int i = 0; i < 8; ++i) acc[h][j][i] = 0.0f;

#pragma unroll
    for (int k0 = 0; k0 < K; k0 += 32) {
        v16bf afrag0 = load_frag(arow0 + k0);
        v16bf afrag1 = load_frag(arow1 + k0);
        v16bf bfrag[4];
#pragma unroll
        for (int j = 0; j < 4; ++j)
            bfrag[j] = load_frag(brow + (size_t)j * 16 * K + k0);
#pragma unroll
        for (int j = 0; j < 4; ++j)
            acc[0][j] = __builtin_amdgcn_wmma_f32_16x16x32_bf16(
                false, afrag0, false, bfrag[j], (short)0, acc[0][j], false, false);
#pragma unroll
        for (int j = 0; j < 4; ++j)
            acc[1][j] = __builtin_amdgcn_wmma_f32_16x16x32_bf16(
                false, afrag1, false, bfrag[j], (short)0, acc[1][j], false, false);
    }

#pragma unroll
    for (int h = 0; h < 2; ++h) {
        int row_base = tile_m * 32 + h * 16 + (lane >> 4) * 8;
#pragma unroll
        for (int j = 0; j < 4; ++j) {
            int n = tile_n * 64 + j * 16 + lmod;
            float bv = bias[n];
#pragma unroll
            for (int i = 0; i < 8; ++i) {
                float v = acc[h][j][i] + bv;
                if (RELU) v = v > 0.0f ? v : 0.0f;
                size_t idx = (size_t)(row_base + i) * N + n;
                if (OUTBF) Cb[idx] = f2bf(v); else Cf[idx] = v;
            }
        }
    }
}

// ---------------- BatchNorm statistics ----------------
__global__ void zero_f32_kernel(float* __restrict__ p, size_t n) {
    size_t i = (size_t)blockIdx.x * 256 + threadIdx.x;
    if (i < n) p[i] = 0.0f;
}
__global__ __launch_bounds__(256) void bn_stats_kernel(
    const float* __restrict__ z, float* __restrict__ stats, int nrows) {
    int c = threadIdx.x;
    int r0 = blockIdx.x * 128;
    int rend = r0 + 128; if (rend > nrows) rend = nrows;
    float s = 0.0f, s2 = 0.0f;
    for (int r = r0; r < rend; ++r) {
        float v = z[(size_t)r * HID + c];
        s += v; s2 += v * v;
    }
    atomicAdd(&stats[c], s);
    atomicAdd(&stats[HID + c], s2);
}
__global__ void bn_finalize_kernel(float* __restrict__ stats,
                                   const float* __restrict__ gammas,
                                   const float* __restrict__ betas, int layer) {
    int c = threadIdx.x;
    const float invN = 1.0f / (float)N_NODES;
    float mu = stats[c] * invN;
    float var = stats[HID + c] * invN - mu * mu;
    float sc = gammas[layer * HID + c] * rsqrtf(var + 1e-5f);
    stats[2 * HID + c] = sc;
    stats[3 * HID + c] = betas[layer * HID + c] - mu * sc;
}
__global__ __launch_bounds__(256) void bn_apply_kernel(
    const float* __restrict__ z, const float* __restrict__ stats,
    float* __restrict__ h) {
    size_t i = (size_t)blockIdx.x * 256 + threadIdx.x;
    int c = (int)(i & (HID - 1));
    float v = z[i] * stats[2 * HID + c] + stats[3 * HID + c];
    h[i] = v > 0.0f ? v : 0.0f;
}

// ---------------- segment mean pool ----------------
__global__ __launch_bounds__(256) void pool_scatter_kernel(
    const float* __restrict__ h, const int* __restrict__ batch,
    float* __restrict__ pooled, float* __restrict__ counts) {
    int node = blockIdx.x;
    int g = batch[node];
    int c = threadIdx.x;
    atomicAdd(&pooled[(size_t)g * HID + c], h[(size_t)node * HID + c]);
    if (c == 0) atomicAdd(&counts[g], 1.0f);
}
__global__ __launch_bounds__(256) void pool_div_kernel(
    float* __restrict__ pooled, const float* __restrict__ counts) {
    int i = blockIdx.x * 256 + threadIdx.x;
    float cnt = counts[i >> 8];
    pooled[i] /= (cnt < 1.0f ? 1.0f : cnt);
}

// ---------------- final matvec: out[g] = relu_t[g,:] . Wr2 + br2 ----------
__global__ __launch_bounds__(256) void readout2_kernel(
    const float* __restrict__ t, const float* __restrict__ Wr2,
    const float* __restrict__ br2, float* __restrict__ out) {
    __shared__ float red[256];
    int g = blockIdx.x, c = threadIdx.x;
    red[c] = t[(size_t)g * HID + c] * Wr2[c];
    __syncthreads();
    for (int s = 128; s > 0; s >>= 1) {
        if (c < s) red[c] += red[c + s];
        __syncthreads();
    }
    if (c == 0) out[g] = red[0] + br2[0];
}

extern "C" void kernel_launch(void* const* d_in, const int* in_sizes, int n_in,
                              void* d_out, int out_size, void* d_ws, size_t ws_size,
                              hipStream_t stream) {
    const float* x    = (const float*)d_in[0];
    const int*   ei   = (const int*)d_in[1];
    const int*   batch= (const int*)d_in[2];
    const float* Wp   = (const float*)d_in[3];
    const float* bp   = (const float*)d_in[4];
    const float* eps  = (const float*)d_in[5];
    const float* W1s  = (const float*)d_in[6];
    const float* b1s  = (const float*)d_in[7];
    const float* W2s  = (const float*)d_in[8];
    const float* b2s  = (const float*)d_in[9];
    const float* gammas = (const float*)d_in[10];
    const float* betas  = (const float*)d_in[11];
    const float* Wr1  = (const float*)d_in[12];
    const float* br1  = (const float*)d_in[13];
    const float* Wr2  = (const float*)d_in[14];
    const float* br2  = (const float*)d_in[15];
    float* out = (float*)d_out;

    const size_t NH = (size_t)N_NODES * HID;            // 25,600,000
    char* wsb = (char*)d_ws;
    float*  hA     = (float*)wsb;                       wsb += NH * 4;
    float*  hB     = (float*)wsb;                       wsb += NH * 4;
    __bf16* abf    = (__bf16*)wsb;                      wsb += NH * 2;
    __bf16* abf2   = (__bf16*)wsb;                      wsb += NH * 2;
    float*  stats  = (float*)wsb;                       wsb += 4 * HID * 4;
    float*  pooled = (float*)wsb;                       wsb += (size_t)N_GRAPHS * HID * 4;
    float*  counts = (float*)wsb;                       wsb += (size_t)N_GRAPHS * 4;
    float*  tbuf   = (float*)wsb;                       wsb += (size_t)N_GRAPHS * HID * 4;
    __bf16* pbf    = (__bf16*)wsb;                      wsb += (size_t)N_GRAPHS * HID * 2;
    __bf16* wT     = (__bf16*)wsb;                      wsb += (size_t)HID * HID * 2;

    const int EW_BLOCKS = (int)(NH / 256);              // 100000
    const int MT32      = N_NODES / 32;                 // 3125 (exact)
    const int GEMM_BLOCKS = (MT32 * 4 + 7) / 8;         // 1563
    const int RO_MT32   = N_GRAPHS / 32;                // 64
    const int RO_BLOCKS = (RO_MT32 * 4) / 8;            // 32

    // h = x @ Wp + bp
    proj_kernel<<<N_NODES / 4, 256, 0, stream>>>(x, Wp, bp, hA);

    for (int l = 0; l < LAYERS; ++l) {
        // z = (1+eps)h + scatter_add(h[src] -> dst)
        gin_init_kernel<<<EW_BLOCKS, 256, 0, stream>>>(hA, hB, eps, l);
        edge_scatter_kernel<<<N_EDGES, 256, 0, stream>>>(ei, hA, hB);
        // MLP GEMM 1 (relu, bf16 out feeds GEMM 2)
        cvt_bf16_kernel<<<EW_BLOCKS, 256, 0, stream>>>(hB, abf);
        wconvT_kernel<<<HID, HID, 0, stream>>>(W1s + (size_t)l * HID * HID, wT);
        gemm_bf16_wmma<true, true><<<GEMM_BLOCKS, 256, 0, stream>>>(
            abf, wT, b1s + l * HID, nullptr, abf2, MT32);
        // MLP GEMM 2 (f32 out for BN)
        wconvT_kernel<<<HID, HID, 0, stream>>>(W2s + (size_t)l * HID * HID, wT);
        gemm_bf16_wmma<false, false><<<GEMM_BLOCKS, 256, 0, stream>>>(
            abf2, wT, b2s + l * HID, hB, nullptr, MT32);
        // BatchNorm (training stats) + relu -> hA
        zero_f32_kernel<<<2, 256, 0, stream>>>(stats, 2 * HID);
        bn_stats_kernel<<<(N_NODES + 127) / 128, 256, 0, stream>>>(hB, stats, N_NODES);
        bn_finalize_kernel<<<1, HID, 0, stream>>>(stats, gammas, betas, l);
        bn_apply_kernel<<<EW_BLOCKS, 256, 0, stream>>>(hB, stats, hA);
    }

    // global mean pool
    zero_f32_kernel<<<((size_t)N_GRAPHS * (HID + 1) + 255) / 256, 256, 0, stream>>>(
        pooled, (size_t)N_GRAPHS * (HID + 1));   // pooled + counts are contiguous
    pool_scatter_kernel<<<N_NODES, 256, 0, stream>>>(hA, batch, pooled, counts);
    pool_div_kernel<<<N_GRAPHS, 256, 0, stream>>>(pooled, counts);

    // readout MLP
    cvt_bf16_kernel<<<(N_GRAPHS * HID) / 256, 256, 0, stream>>>(pooled, pbf);
    wconvT_kernel<<<HID, HID, 0, stream>>>(Wr1, wT);
    gemm_bf16_wmma<true, false><<<RO_BLOCKS, 256, 0, stream>>>(
        pbf, wT, br1, tbuf, nullptr, RO_MT32);
    readout2_kernel<<<N_GRAPHS, 256, 0, stream>>>(tbuf, Wr2, br2, out);
}
